// NaiveConvolutionalLayer_72026601554521
// MI455X (gfx1250) — matrix operand order repaired and checked by reading, not compiled, not run on hardware
//
#include <hip/hip_runtime.h>
#include <hip/hip_bf16.h>

#define N_NODES 50000
#define N_EDGES 1600000
#define F_NODE  64
#define F_EDGE  32
#define MSG     32

typedef __attribute__((ext_vector_type(16))) __bf16 v16bf;
typedef __attribute__((ext_vector_type(8)))  float  v8f;

union BFfrag {
    v16bf          v;
    unsigned short u[16];
    uint4          q[2];
};

// f32 -> bf16 via hardware convert (lowers to v_cvt_pk_bf16_f32 class ops, RNE)
__device__ __forceinline__ __bf16 f2bf(float f) { return (__bf16)f; }

__global__ __launch_bounds__(256) void zero_f4(float4* __restrict__ p, int n4) {
    int i = blockIdx.x * blockDim.x + threadIdx.x;
    if (i < n4) p[i] = make_float4(0.f, 0.f, 0.f, 0.f);
}

// Convert 4 f32 (float4) into bf16 fragment elements [base..base+3]
__device__ __forceinline__ void cvt4(BFfrag& a, int base, float4 q) {
    a.v[base + 0] = f2bf(q.x);
    a.v[base + 1] = f2bf(q.y);
    a.v[base + 2] = f2bf(q.z);
    a.v[base + 3] = f2bf(q.w);
}

// ---------------------------------------------------------------------------
// Edge stage: messages = relu([nf[n0], nf[n1], ef] @ W_edge + b_edge)
// then atomic scatter-add into msum[node0].
// One wave = 16 edges x 32 msg outputs. K=160 -> 5 chunks of 32,
// each chunk entirely within one gather source.
// ---------------------------------------------------------------------------
__global__ __launch_bounds__(256) void edge_msg_kernel(
    const float* __restrict__ nf, const int* __restrict__ eidx,
    const float* __restrict__ ef, const float* __restrict__ We,
    const float* __restrict__ be, float* __restrict__ msum)
{
    // W_edge staged in LDS pre-swizzled into B-fragment order:
    // [kchunk][ntile][lane][j], j contiguous -> 2x 16B DS loads per fragment.
    __shared__ unsigned short sWe[5][2][32][16];
    __shared__ float sBe[MSG];

    const int tid = threadIdx.x;
    for (int i = tid; i < 5 * 2 * 32 * 16; i += 256) {
        int j  = i & 15;
        int ln = (i >> 4) & 31;
        int t  = (i >> 9) & 1;
        int c  = i >> 10;
        int K  = c * 32 + ((ln >> 4) << 3) + (j & 7) + ((j >> 3) << 4);   // half-wave holds K{8h..8h+7, 16+8h..16+8h+7}
        int n  = t * 16 + (ln & 15);
        __bf16 h = f2bf(We[K * MSG + n]);
        sWe[c][t][ln][j] = __builtin_bit_cast(unsigned short, h);
    }
    if (tid < MSG) sBe[tid] = be[tid];
    __syncthreads();

    const int lane  = tid & 31;
    const int wave  = tid >> 5;
    const int hi    = lane >> 4;   // 0: lanes 0-15, 1: lanes 16-31
    const int mrow  = lane & 15;
    const int ebase = blockIdx.x * 128 + wave * 16;
    if (ebase >= N_EDGES) return;  // N_EDGES % 128 == 0 -> tiles always full

    const int e  = ebase + mrow;
    const int n0 = eidx[e];
    const int n1 = eidx[N_EDGES + e];

    // next tile's edge-feature stream prefetch
    if (e + 128 < N_EDGES)
        __builtin_prefetch(ef + (size_t)(e + 128) * F_EDGE, 0, 0);

    const float* src[5];
    src[0] = nf + (size_t)n0 * F_NODE;       // K   0..31
    src[1] = src[0] + 32;                    // K  32..63
    src[2] = nf + (size_t)n1 * F_NODE;       // K  64..95
    src[3] = src[2] + 32;                    // K  96..127
    src[4] = ef + (size_t)e * F_EDGE;        // K 128..159

    v8f acc0 = {}, acc1 = {};

#pragma unroll
    for (int c = 0; c < 5; ++c) {
        // A fragment, 16-bit A layout: lane<16 K{0..7,16..23}, lane>=16 K{8..15,24..31}
        const float* rowp = src[c] + hi * 8;
        float4 q0 = *(const float4*)(rowp + 0);
        float4 q1 = *(const float4*)(rowp + 4);
        float4 q2 = *(const float4*)(rowp + 16);
        float4 q3 = *(const float4*)(rowp + 20);
        BFfrag a;
        cvt4(a, 0, q0);
        cvt4(a, 4, q1);
        cvt4(a, 8, q2);
        cvt4(a, 12, q3);

        BFfrag b0, b1;
        const uint4* bp0 = (const uint4*)&sWe[c][0][lane][0];
        const uint4* bp1 = (const uint4*)&sWe[c][1][lane][0];
        b0.q[0] = bp0[0]; b0.q[1] = bp0[1];
        b1.q[0] = bp1[0]; b1.q[1] = bp1[1];

        acc0 = __builtin_amdgcn_wmma_f32_16x16x32_bf16(false, a.v, false, b0.v,
                                                       (short)0, acc0, false, false);
        acc1 = __builtin_amdgcn_wmma_f32_16x16x32_bf16(false, a.v, false, b1.v,
                                                       (short)0, acc1, false, false);
    }

    // bias + relu + scatter-sum (hardware f32 atomics; msum is L2-resident)
    const int nlo = lane & 15;
#pragma unroll
    for (int r = 0; r < 8; ++r) {
        const int erow = ebase + r + hi * 8;   // edge row for C/D VGPR r
        const int dn   = eidx[erow];           // node0 of that edge
        float v0 = acc0[r] + sBe[nlo];
        float v1 = acc1[r] + sBe[16 + nlo];
        v0 = v0 > 0.f ? v0 : 0.f;
        v1 = v1 > 0.f ? v1 : 0.f;
        unsafeAtomicAdd(&msum[(size_t)dn * MSG + nlo], v0);
        unsafeAtomicAdd(&msum[(size_t)dn * MSG + 16 + nlo], v1);
    }
}

// ---------------------------------------------------------------------------
// Node stage: out = relu([nf, msum] @ W_node + b_node)
// One wave = 16 nodes x 64 outputs. K=96 -> 3 chunks; N=64 -> 4 tiles.
// ---------------------------------------------------------------------------
__global__ __launch_bounds__(256) void node_update_kernel(
    const float* __restrict__ nf, const float* __restrict__ msum,
    const float* __restrict__ Wn, const float* __restrict__ bn,
    float* __restrict__ out)
{
    __shared__ unsigned short sWn[3][4][32][16];
    __shared__ float sBn[F_NODE];

    const int tid = threadIdx.x;
    for (int i = tid; i < 3 * 4 * 32 * 16; i += 256) {
        int j  = i & 15;
        int ln = (i >> 4) & 31;
        int t  = (i >> 9) & 3;
        int c  = i >> 11;
        int K  = c * 32 + ((ln >> 4) << 3) + (j & 7) + ((j >> 3) << 4);
        int n  = t * 16 + (ln & 15);
        __bf16 h = f2bf(Wn[K * F_NODE + n]);
        sWn[c][t][ln][j] = __builtin_bit_cast(unsigned short, h);
    }
    if (tid < F_NODE) sBn[tid] = bn[tid];
    __syncthreads();

    const int lane = tid & 31;
    const int wave = tid >> 5;
    const int hi   = lane >> 4;
    const int mrow = lane & 15;
    const int nb   = blockIdx.x * 128 + wave * 16;
    if (nb >= N_NODES) return;   // N_NODES % 16 == 0 -> active tiles are full

    const int node = nb + mrow;
    const float* src[3];
    src[0] = nf + (size_t)node * F_NODE;      // K  0..31
    src[1] = src[0] + 32;                     // K 32..63
    src[2] = msum + (size_t)node * MSG;       // K 64..95

    v8f acc[4] = {{}, {}, {}, {}};

#pragma unroll
    for (int c = 0; c < 3; ++c) {
        const float* rowp = src[c] + hi * 8;
        float4 q0 = *(const float4*)(rowp + 0);
        float4 q1 = *(const float4*)(rowp + 4);
        float4 q2 = *(const float4*)(rowp + 16);
        float4 q3 = *(const float4*)(rowp + 20);
        BFfrag a;
        cvt4(a, 0, q0);
        cvt4(a, 4, q1);
        cvt4(a, 8, q2);
        cvt4(a, 12, q3);

#pragma unroll
        for (int t = 0; t < 4; ++t) {
            BFfrag b;
            const uint4* bp = (const uint4*)&sWn[c][t][lane][0];
            b.q[0] = bp[0]; b.q[1] = bp[1];
            acc[t] = __builtin_amdgcn_wmma_f32_16x16x32_bf16(false, a.v, false, b.v,
                                                             (short)0, acc[t], false, false);
        }
    }

#pragma unroll
    for (int t = 0; t < 4; ++t) {
        const int col = t * 16 + (lane & 15);
        const float bv = sBn[col];
#pragma unroll
        for (int r = 0; r < 8; ++r) {
            const int row = nb + r + hi * 8;
            float v = acc[t][r] + bv;
            out[(size_t)row * F_NODE + col] = v > 0.f ? v : 0.f;
        }
    }
}

extern "C" void kernel_launch(void* const* d_in, const int* in_sizes, int n_in,
                              void* d_out, int out_size, void* d_ws, size_t ws_size,
                              hipStream_t stream) {
    const float* nf   = (const float*)d_in[0];  // [N, 64]
    const int*   eidx = (const int*)  d_in[1];  // [2, E]
    const float* ef   = (const float*)d_in[2];  // [E, 32]
    const float* We   = (const float*)d_in[3];  // [160, 32]
    const float* be   = (const float*)d_in[4];  // [32]
    const float* Wn   = (const float*)d_in[5];  // [96, 64]
    const float* bn   = (const float*)d_in[6];  // [64]
    float* out  = (float*)d_out;                // [N, 64]
    float* msum = (float*)d_ws;                 // [N, 32] scratch (6.4 MB)

    const int n4 = N_NODES * MSG / 4;
    zero_f4<<<(n4 + 255) / 256, 256, 0, stream>>>((float4*)msum, n4);

    edge_msg_kernel<<<(N_EDGES + 127) / 128, 256, 0, stream>>>(
        nf, eidx, ef, We, be, msum);

    node_update_kernel<<<(N_NODES + 127) / 128, 256, 0, stream>>>(
        nf, msum, Wn, bn, out);
}
